// CCNLoss_77232101916940
// MI455X (gfx1250) — matrix operand order repaired and checked by reading, not compiled, not run on hardware
//
#include <hip/hip_runtime.h>

// ---------------- CDNA5 WMMA / TDM types ----------------
typedef __attribute__((ext_vector_type(16))) __bf16 v16bf;
typedef __attribute__((ext_vector_type(8)))  float  v8f;
typedef unsigned int v4u  __attribute__((ext_vector_type(4)));
typedef int          v4i_t __attribute__((ext_vector_type(4)));
typedef int          v8i_t __attribute__((ext_vector_type(8)));

// Problem constants (B=4, C=3, H=W=1024)
#define HW    1048576      // H*W
#define CHW   3145728      // C*H*W
#define BCHW  12582912     // B*C*H*W
#define N3    3072         // C*H  (GEMM M and N)
#define KDIM  1024         // GEMM K

// ---------------- helpers ----------------
__device__ inline unsigned short f2bf(float f) {
  unsigned int u = __float_as_uint(f);
  unsigned int r = u + 0x7FFFu + ((u >> 16) & 1u);   // round-to-nearest-even
  return (unsigned short)(r >> 16);
}

__device__ inline float block_reduce_256(float v, float* sm) {
  int t = threadIdx.x;
  sm[t] = v;
  __syncthreads();
  for (int s = 128; s > 0; s >>= 1) {
    if (t < s) sm[t] += sm[t + s];
    __syncthreads();
  }
  float r = sm[0];
  __syncthreads();
  return r;
}

// ---------------- fp32 -> bf16 conversion ----------------
__global__ void cvt_bf16_kernel(const float* __restrict__ src,
                                unsigned short* __restrict__ dst, int n) {
  int i = blockIdx.x * blockDim.x + threadIdx.x;
  int stride = gridDim.x * blockDim.x;
  for (; i < n; i += stride) dst[i] = f2bf(src[i]);
}

// ---------------- fused r_loss + c_loss (single full-tensor pass) ----------------
__global__ void __launch_bounds__(256)
rc_loss_kernel(const float* __restrict__ p, const float* __restrict__ t,
               float* __restrict__ partial_r, float* __restrict__ partial_c) {
  int pix = blockIdx.x * 256 + threadIdx.x;          // grid = 4096 -> exactly HW
  float rsum = 0.0f, cosv = 0.0f;
#pragma unroll
  for (int c = 0; c < 3; ++c) {
    float dot = 0.0f, na = 0.0f, nb = 0.0f;
#pragma unroll
    for (int b = 0; b < 4; ++b) {
      float x = p[b * CHW + c * HW + pix];
      float y = t[b * CHW + c * HW + pix];
      rsum += fabsf(x - y);
      dot += x * y; na += x * x; nb += y * y;
    }
    cosv += dot / (fmaxf(sqrtf(na), 1e-12f) * fmaxf(sqrtf(nb), 1e-12f));
  }
  cosv = fminf(fmaxf(cosv, -1.0f + 1e-7f), 1.0f - 1e-7f);
  float cl = acosf(cosv);

  __shared__ float sm[256];
  float r  = block_reduce_256(rsum, sm);
  float cc = block_reduce_256(cl,   sm);
  if (threadIdx.x == 0) {
    partial_r[blockIdx.x] = r;
    partial_c[blockIdx.x] = cc;
  }
}

// ---------------- bf16 WMMA GEMM: X[m,n] = sum_k Q[k,m] * P[k,n] ----------------
#define BK   32
#define ASTR 40   // ushorts per LDS A row (32 + 8 pad) -> 80B stride, 16B aligned
#define BSTR 72   // ushorts per LDS B row (64 + 8 pad) -> 144B stride, 16B aligned
                  // (TDM pad: every 32 DWORDs insert 4 DWORDs -> same 72-ushort rows)

__global__ void __launch_bounds__(128)
gemm_bf16_kernel(const unsigned short* __restrict__ Pbf,
                 const unsigned short* __restrict__ Qbf,
                 float* __restrict__ X) {
  __shared__ unsigned short As[64 * ASTR];  // As[m][k] = Q[k][m0+m]  (transposed stage)
  __shared__ unsigned short Bs[32 * BSTR];  // Bs[k][n] = P[k][n0+n]  (staged by TDM)

  const int tid  = threadIdx.x;
  const int lane = tid & 31;
  const int wid  = tid >> 5;                // 0..3 (wave32)
  const int wm   = wid >> 1, wn = wid & 1;  // 2x2 wave grid, each wave 32x32
  const int m0   = blockIdx.y * 64;
  const int n0   = blockIdx.x * 64;
  const int h    = lane >> 4;               // half-wave select
  const int l16  = lane & 15;

  v8f acc[2][2] = {};

  const unsigned int* Q32 = reinterpret_cast<const unsigned int*>(Qbf);
  const unsigned long long pbase = (unsigned long long)(uintptr_t)Pbf;
  const unsigned int bs_lds = (unsigned int)(uintptr_t)&Bs[0]; // LDS byte offset

  for (int kt = 0; kt < KDIM; kt += BK) {
    // ---- B tile via Tensor Data Mover: 32x64 bf16 tile of P, padded rows ----
    if (tid < 32) {                          // wave 0 issues one TDM op (EXEC ignored)
      unsigned long long ga =
          pbase + (unsigned long long)((unsigned)(kt * N3 + n0) * 2u);
      v4u g0;
      g0.x = 1u;                                         // count=1, user mode
      g0.y = bs_lds;                                     // lds_addr (bytes)
      g0.z = (unsigned int)(ga & 0xFFFFFFFFull);         // global_addr[31:0]
      g0.w = (unsigned int)(ga >> 32) | (2u << 30);      // addr[56:32] | type=2
      v8i_t g1;
      g1[0] = 0x07110000;   // data_size=2B | pad_enable | pad_interval=32dw | pad=4dw
      g1[1] = 0x0C000000;   // tensor_dim0 = 3072 (low16 << 16)
      g1[2] = 0x04000000;   // tensor_dim0 hi | tensor_dim1 = 1024 (low16 << 16)
      g1[3] = 0x00400000;   // tensor_dim1 hi | tile_dim0 = 64
      g1[4] = 32;           // tile_dim1 = 32, tile_dim2 = 0
      g1[5] = 3072;         // tensor_dim0_stride (low 32)
      g1[6] = 0;            // stride0 hi | stride1 lo
      g1[7] = 0;            // stride1 hi
      v4i_t z4 = (v4i_t)0;
#if defined(__clang_major__) && (__clang_major__ >= 23)
      v8i_t z8 = (v8i_t)0;
      __builtin_amdgcn_tensor_load_to_lds(g0, g1, z4, z4, z8, 0);
#else
      __builtin_amdgcn_tensor_load_to_lds(g0, g1, z4, z4, 0);
#endif
      __builtin_amdgcn_s_wait_tensorcnt(0);
    }

    // ---- A tile staged manually (needs k<->m transpose into LDS) ----
#pragma unroll
    for (int i = 0; i < 8; ++i) {
      int idx = i * 128 + tid;              // 0..1023 uints (32 rows x 32 uints)
      int kr  = idx >> 5;
      int cp  = idx & 31;
      unsigned int qa = Q32[((kt + kr) * N3 + m0) / 2 + cp];
      As[(cp * 2)     * ASTR + kr] = (unsigned short)(qa & 0xFFFFu);
      As[(cp * 2 + 1) * ASTR + kr] = (unsigned short)(qa >> 16);
    }
    if (kt + BK < KDIM)                     // prefetch next A K-strip
      __builtin_prefetch(&Q32[((kt + BK) * N3 + m0) / 2 + tid], 0, 1);
    __syncthreads();

    // ---- fragments per documented 16-bit VGPR layouts ----
    union FragU { v16bf v; uint4 q[2]; };
    FragU a[2], b[2];
#pragma unroll
    for (int mi = 0; mi < 2; ++mi) {
      int row = (wm * 2 + mi) * 16 + l16;   // A: M = lane%16; K chunks 8h.., 16+8h..
      const unsigned short* rp = &As[row * ASTR];
      a[mi].q[0] = *reinterpret_cast<const uint4*>(rp + 8 * h);
      a[mi].q[1] = *reinterpret_cast<const uint4*>(rp + 16 + 8 * h);
    }
#pragma unroll
    for (int ni = 0; ni < 2; ++ni) {
      int nsub = (wn * 2 + ni) * 16;        // B: N = lane%16; K chunks 8h.., 16+8h..
      const unsigned short* rp = &Bs[nsub + l16];
#pragma unroll
      for (int s = 0; s < 8; ++s) {
        b[ni].v[s]     = __builtin_bit_cast(__bf16, rp[(8 * h + s) * BSTR]);
        b[ni].v[8 + s] = __builtin_bit_cast(__bf16, rp[(16 + 8 * h + s) * BSTR]);
      }
    }

#pragma unroll
    for (int mi = 0; mi < 2; ++mi)
#pragma unroll
      for (int ni = 0; ni < 2; ++ni)
        acc[mi][ni] = __builtin_amdgcn_wmma_f32_16x16x32_bf16(
            false, a[mi].v, false, b[ni].v, (short)0, acc[mi][ni], false, false);
    __syncthreads();
  }

  // ---- store C/D: lane -> N = lane%16; VGPR r -> M = r + 8*(lane/16) ----
#pragma unroll
  for (int mi = 0; mi < 2; ++mi)
#pragma unroll
    for (int ni = 0; ni < 2; ++ni) {
      int col = n0 + (wn * 2 + ni) * 16 + l16;
#pragma unroll
      for (int r = 0; r < 8; ++r) {
        int rowg = m0 + (wm * 2 + mi) * 16 + r + 8 * h;
        X[rowg * N3 + col] = acc[mi][ni][r];
      }
    }
}

// ---------------- column softmax -> mean contribution ----------------
__global__ void __launch_bounds__(256)
colsoftmax_kernel(const float* __restrict__ X, float* __restrict__ partial) {
  int n = blockIdx.x * 256 + threadIdx.x;   // 12 blocks -> exactly 3072 columns
  float mx = -3.402823466e38f;
  for (int m = 0; m < N3; ++m) mx = fmaxf(mx, X[m * N3 + n]);
  float s = 0.0f;
  for (int m = 0; m < N3; ++m) s += expf(X[m * N3 + n] - mx);
  float tot = 0.0f;
  for (int m = 0; m < N3; ++m) tot += expf(X[m * N3 + n] - mx) / s;
  __shared__ float sm[256];
  float r = block_reduce_256(tot, sm);
  if (threadIdx.x == 0) partial[blockIdx.x] = r;
}

// ---------------- final combine ----------------
__global__ void __launch_bounds__(256)
finalize_kernel(const float* __restrict__ pr, const float* __restrict__ pc,
                const float* __restrict__ smp, const float* __restrict__ smt,
                float* __restrict__ out) {
  __shared__ float sm[256];
  int t = threadIdx.x;
  float sr = 0.0f, sc = 0.0f;
  for (int i = t; i < 4096; i += 256) { sr += pr[i]; sc += pc[i]; }
  sr = block_reduce_256(sr, sm);
  sc = block_reduce_256(sc, sm);
  if (t == 0) {
    float sp = 0.0f, st = 0.0f;
    for (int i = 0; i < 12; ++i) { sp += smp[i]; st += smt[i]; }
    const float NN = (float)N3 * (float)N3;
    float cr = fabsf(sp / NN - st / NN);
    out[0] = sr / (float)BCHW + 0.5f * sc + 1.0f * cr;
  }
}

// ---------------- launcher ----------------
extern "C" void kernel_launch(void* const* d_in, const int* in_sizes, int n_in,
                              void* d_out, int out_size, void* d_ws, size_t ws_size,
                              hipStream_t stream) {
  const float* pred = (const float*)d_in[0];
  const float* tgt  = (const float*)d_in[1];
  float* out = (float*)d_out;

  // workspace layout
  float* wsf = (float*)d_ws;
  float* pr  = wsf;                 // 4096 floats
  float* pc  = wsf + 4096;          // 4096 floats
  float* smp = wsf + 8192;          // 12 floats
  float* smt = wsf + 8204;          // 12 floats
  unsigned short* bfPred = (unsigned short*)((char*)d_ws + 65536);       // 2*CHW bf16
  unsigned short* bfTgt  = bfPred + 2 * CHW;                             // 2*CHW bf16
  float* Xbuf = (float*)((char*)d_ws + 65536 + (size_t)8 * CHW);         // 3072^2 fp32

  // fp32 -> bf16 (img0 = P, img1 = Q for each tensor)
  cvt_bf16_kernel<<<4096, 256, 0, stream>>>(pred, bfPred, 2 * CHW);
  cvt_bf16_kernel<<<4096, 256, 0, stream>>>(tgt,  bfTgt,  2 * CHW);

  // fused r_loss + c_loss (single full read of both tensors)
  rc_loss_kernel<<<4096, 256, 0, stream>>>(pred, tgt, pr, pc);

  // channel relevance: GEMM (WMMA bf16, TDM-staged B tiles) + column softmax mean
  dim3 gg(N3 / 64, N3 / 64);
  gemm_bf16_kernel<<<gg, 128, 0, stream>>>(bfPred, bfPred + CHW, Xbuf);
  colsoftmax_kernel<<<12, 256, 0, stream>>>(Xbuf, smp);
  gemm_bf16_kernel<<<gg, 128, 0, stream>>>(bfTgt, bfTgt + CHW, Xbuf);
  colsoftmax_kernel<<<12, 256, 0, stream>>>(Xbuf, smt);

  finalize_kernel<<<1, 256, 0, stream>>>(pr, pc, smp, smt, out);
}
